// TEMixtralSparseMoeBlock_7000796692531
// MI455X (gfx1250) — matrix operand order repaired and checked by reading, not compiled, not run on hardware
//
#include <hip/hip_runtime.h>
#include <hip/hip_bf16.h>

// ---------------- problem constants ----------------
#define HIDDEN 1024
#define FFN    3584
#define NE     8
#define T_TOK  2048          // B*S
#define NSLOT  (T_TOK * 2)   // total top-2 assignments = 4096 (exact)

typedef unsigned short u16;
typedef unsigned int   u32;

typedef __attribute__((ext_vector_type(16))) __bf16 v16bf;
typedef __attribute__((ext_vector_type(8)))  __bf16 v8bf;
typedef __attribute__((ext_vector_type(8)))  float  v8f;

// ---- Tensor Data Mover availability (this toolchain: 6-arg builtin) ----
#define USE_TDM 0
#if defined(__has_builtin)
#if __has_builtin(__builtin_amdgcn_tensor_load_to_lds) && \
    __has_builtin(__builtin_amdgcn_s_wait_tensorcnt)
#undef USE_TDM
#define USE_TDM 1
#endif
#endif

#if USE_TDM
typedef __attribute__((ext_vector_type(4))) unsigned int u32x4;
typedef __attribute__((ext_vector_type(8))) int          i32x8;
typedef __attribute__((ext_vector_type(4))) int          i32x4;

__device__ __forceinline__ u32 lds_off(const void* p) {
  return (u32)(size_t)(__attribute__((address_space(3))) const void*)p;
}

// 2D TDM load: tile of 64 rows x 32 bf16 (64B) from global (row stride
// `stride_elems` bf16) into LDS at `lds_byte`, padding LDS rows to 80B
// (pad_interval=3 -> every 16 DWORDs, pad_amount=3 -> 4 DWORDs) so the LDS
// layout matches the manual LSTR=40 path. Rows beyond `rows_avail` read as 0.
__device__ __forceinline__ void tdm_load_tile(const u16* gsrc, u32 lds_byte,
                                              u32 stride_elems, u32 rows_avail) {
  unsigned long long ga = (unsigned long long)(size_t)gsrc;
  u32x4 g0;
  g0[0] = 1u;                                           // count=1, user mode
  g0[1] = lds_byte;                                     // lds_addr
  g0[2] = (u32)ga;                                      // global_addr[31:0]
  g0[3] = ((u32)(ga >> 32) & 0x01FFFFFFu) | (2u << 30); // addr[56:32] | type=2
  i32x8 g1;
  g1[0] = (int)((1u << 16)    // data_size = 2 bytes
              | (1u << 20)    // pad_enable
              | (3u << 22)    // pad_interval = 16 DWORDs (64B)
              | (3u << 25));  // pad_amount   = 4 DWORDs (16B)
  g1[1] = (int)(32u << 16);                       // tensor_dim0 = 32 (lo16)
  g1[2] = (int)((rows_avail & 0xFFFFu) << 16);    // tensor_dim1[15:0]
  g1[3] = (int)((rows_avail >> 16) | (32u << 16)); // dim1[31:16] | tile_dim0=32
  g1[4] = (int)64u;                               // tile_dim1 = 64 rows
  g1[5] = (int)stride_elems;                      // tensor_dim0_stride[31:0]
  g1[6] = 0;
  g1[7] = 0;
  i32x4 z4 = {0, 0, 0, 0};
  i32x8 z8 = {0, 0, 0, 0, 0, 0, 0, 0};
  __builtin_amdgcn_tensor_load_to_lds(g0, g1, z4, z4, z8, 0);
}
#endif

// fp32 -> bf16 round-to-nearest-even
__device__ __forceinline__ u16 f2bf(float f) {
  u32 u = __float_as_uint(f);
  u32 r = (u + 0x7FFFu + ((u >> 16) & 1u)) >> 16;
  return (u16)r;
}

// Build a v16bf A/B fragment from two 16B-aligned LDS chunks.
__device__ __forceinline__ v16bf ld_frag(const u16* p0, const u16* p1) {
  v8bf lo = *(const v8bf*)p0;
  v8bf hi = *(const v8bf*)p1;
  v16bf r;
#pragma unroll
  for (int i = 0; i < 8; ++i) { r[i] = lo[i]; r[i + 8] = hi[i]; }
  return r;
}

__device__ __forceinline__ v8f wmma_bf16(v16bf a, v16bf b, v8f c) {
  // (neg_a, A, neg_b, B, c_mod, C, reuse_a, reuse_b)
  return __builtin_amdgcn_wmma_f32_16x16x32_bf16(false, a, false, b, (short)0, c,
                                                 false, false);
}

// ---------------- kernel 0: zero expert counters ----------------
__global__ void k_init(int* cnt) {
  if (threadIdx.x < NE) cnt[threadIdx.x] = 0;
}

// ---------------- kernel 1: router (one wave32 per token) ----------------
__global__ __launch_bounds__(256) void k_router(
    const float* __restrict__ x, const float* __restrict__ gw,
    float* __restrict__ logits_out, int* __restrict__ cnt,
    int* __restrict__ tok_e, int* __restrict__ tok_pos, float* __restrict__ tok_w) {
  const int lane = threadIdx.x & 31;
  const int wv   = threadIdx.x >> 5;
  const int t    = blockIdx.x * 8 + wv;
  const float* xr = x + (size_t)t * HIDDEN;

  float acc[NE];
#pragma unroll
  for (int e = 0; e < NE; ++e) acc[e] = 0.f;

  for (int i = 0; i < HIDDEN / 32; ++i) {
    int idx = lane + i * 32;
    float xv = xr[idx];
#pragma unroll
    for (int e = 0; e < NE; ++e) acc[e] += xv * gw[e * HIDDEN + idx];
  }
#pragma unroll
  for (int e = 0; e < NE; ++e) {
#pragma unroll
    for (int o = 16; o > 0; o >>= 1) acc[e] += __shfl_xor(acc[e], o, 32);
  }

  if (lane == 0) {
#pragma unroll
    for (int e = 0; e < NE; ++e) logits_out[t * NE + e] = acc[e];
    // softmax denominator cancels in top-2 renormalization: use exp(l-m)
    float m = acc[0];
#pragma unroll
    for (int e = 1; e < NE; ++e) m = fmaxf(m, acc[e]);
    float p[NE];
#pragma unroll
    for (int e = 0; e < NE; ++e) p[e] = __expf(acc[e] - m);
    int e0 = 0; float b0 = p[0];
#pragma unroll
    for (int e = 1; e < NE; ++e) if (p[e] > b0) { b0 = p[e]; e0 = e; }
    int e1 = -1; float b1 = -1.f;
#pragma unroll
    for (int e = 0; e < NE; ++e) if (e != e0 && p[e] > b1) { b1 = p[e]; e1 = e; }
    float inv = 1.f / (b0 + b1);
    int p0 = atomicAdd(&cnt[e0], 1);
    int p1 = atomicAdd(&cnt[e1], 1);
    tok_e[t * 2 + 0] = e0; tok_pos[t * 2 + 0] = p0; tok_w[t * 2 + 0] = b0 * inv;
    tok_e[t * 2 + 1] = e1; tok_pos[t * 2 + 1] = p1; tok_w[t * 2 + 1] = b1 * inv;
  }
}

// ---------------- kernel 2: exclusive scan of 8 counters ----------------
__global__ void k_offsets(const int* __restrict__ cnt, int* __restrict__ off) {
  if (threadIdx.x == 0) {
    int s = 0;
    for (int e = 0; e < NE; ++e) { off[e] = s; s += cnt[e]; }
  }
}

// ---------------- kernel 3: gather + fp32->bf16 permuted activations ----------
__global__ __launch_bounds__(256) void k_gather(
    const float* __restrict__ x, const int* __restrict__ tok_e,
    const int* __restrict__ tok_pos, const int* __restrict__ off,
    u16* __restrict__ xp) {
  int a = blockIdx.x;            // assignment id
  int t = a >> 1;
  int slot = off[tok_e[a]] + tok_pos[a];
  int h = threadIdx.x * 4;
  float4 v = *(const float4*)(x + (size_t)t * HIDDEN + h);
  u32 lo = (u32)f2bf(v.x) | ((u32)f2bf(v.y) << 16);
  u32 hi = (u32)f2bf(v.z) | ((u32)f2bf(v.w) << 16);
  *(uint2*)(xp + (size_t)slot * HIDDEN + h) = make_uint2(lo, hi);
}

// ---------------- kernel 4: grouped GEMM1 (gate_up) + SiLU*up -> bf16 act ----
// Block = 256 thr = 8 waves; tile M=64, N=64 (gate) + same 64 cols of up.
// LDS: A[64][K32] row-major, B stored transposed [N64][K32], stride 40 u16.
#define LSTR 40
__global__ __launch_bounds__(256) void k_gemm1(
    const u16* __restrict__ xp, const float* __restrict__ wgu,
    const int* __restrict__ cnt, const int* __restrict__ off,
    u16* __restrict__ act) {
  const int e = blockIdx.z;
  const int count = cnt[e];
  const int M0 = blockIdx.y * 64;
  if (M0 >= count) return;
  const int base = off[e];
  const int n0 = blockIdx.x * 64;

  __shared__ u16 lA[64 * LSTR];
  __shared__ u16 lBg[64 * LSTR];
  __shared__ u16 lBu[64 * LSTR];

  const int tid  = threadIdx.x;
  const int lane = tid & 31;
  const int wv   = tid >> 5;
  const int mi   = wv & 3;      // wave's M tile (0..3)
  const int nj   = wv >> 2;     // wave covers N tiles {2nj, 2nj+1}

  v8f cg0 = {}, cg1 = {}, cu0 = {}, cu1 = {};

#if USE_TDM
  const u32 ldsA = lds_off(lA);
  const u32 rowsAvail = (u32)(NSLOT - (base + M0));
  const u16* atile = xp + (size_t)(base + M0) * HIDDEN;
#else
  const int ar = tid >> 2;                 // 0..63
  const int ac = (tid & 3) * 8;            // 0,8,16,24
  int arow = base + M0 + ar; if (arow > NSLOT - 1) arow = NSLOT - 1;
  const u16* asrc = xp + (size_t)arow * HIDDEN + ac;
#endif

  // B staging: thread reads 8 fp32 of one k-row, writes transposed bf16
  const int bk = tid >> 3;                 // 0..31
  const int bn = (tid & 7) * 8;            // 0..56
  const size_t wbase = (size_t)e * HIDDEN * (2 * FFN);

  const int nlo  = lane & 15;
  const int half = lane >> 4;

  for (int kk = 0; kk < HIDDEN; kk += 32) {
#if USE_TDM
    if (wv == 0) tdm_load_tile(atile + kk, ldsA, HIDDEN, rowsAvail);
#else
    *(uint4*)(lA + ar * LSTR + ac) = *(const uint4*)(asrc + kk);
#endif

    const float* bg = wgu + wbase + (size_t)(kk + bk) * (2 * FFN) + (n0 + bn);
    const float* bu = bg + FFN;
    if (kk + 32 < HIDDEN) {                 // prefetch next K-step weight rows
      __builtin_prefetch(bg + (size_t)32 * (2 * FFN), 0, 1);
      __builtin_prefetch(bu + (size_t)32 * (2 * FFN), 0, 1);
    }
    float4 g0 = *(const float4*)bg, g1 = *(const float4*)(bg + 4);
    float4 u0 = *(const float4*)bu, u1 = *(const float4*)(bu + 4);
    u16* pg = lBg + bk;
    u16* pu = lBu + bk;
    pg[(bn + 0) * LSTR] = f2bf(g0.x); pg[(bn + 1) * LSTR] = f2bf(g0.y);
    pg[(bn + 2) * LSTR] = f2bf(g0.z); pg[(bn + 3) * LSTR] = f2bf(g0.w);
    pg[(bn + 4) * LSTR] = f2bf(g1.x); pg[(bn + 5) * LSTR] = f2bf(g1.y);
    pg[(bn + 6) * LSTR] = f2bf(g1.z); pg[(bn + 7) * LSTR] = f2bf(g1.w);
    pu[(bn + 0) * LSTR] = f2bf(u0.x); pu[(bn + 1) * LSTR] = f2bf(u0.y);
    pu[(bn + 2) * LSTR] = f2bf(u0.z); pu[(bn + 3) * LSTR] = f2bf(u0.w);
    pu[(bn + 4) * LSTR] = f2bf(u1.x); pu[(bn + 5) * LSTR] = f2bf(u1.y);
    pu[(bn + 6) * LSTR] = f2bf(u1.z); pu[(bn + 7) * LSTR] = f2bf(u1.w);

#if USE_TDM
    if (wv == 0) __builtin_amdgcn_s_wait_tensorcnt(0);
#endif
    __syncthreads();

    // load ALL fragments first, then back-to-back WMMAs
    const u16* Ab  = lA  + (mi * 16 + nlo) * LSTR + half * 16;
    const u16* Bg0 = lBg + ((nj * 2) * 16 + nlo) * LSTR + half * 16;
    const u16* Bu0 = lBu + ((nj * 2) * 16 + nlo) * LSTR + half * 16;
    const u16* Bg1 = lBg + ((nj * 2 + 1) * 16 + nlo) * LSTR + half * 16;
    const u16* Bu1 = lBu + ((nj * 2 + 1) * 16 + nlo) * LSTR + half * 16;
    v16bf a    = ld_frag(Ab, Ab + 8);
    v16bf bg0f = ld_frag(Bg0, Bg0 + 8);
    v16bf bu0f = ld_frag(Bu0, Bu0 + 8);
    v16bf bg1f = ld_frag(Bg1, Bg1 + 8);
    v16bf bu1f = ld_frag(Bu1, Bu1 + 8);
    cg0 = wmma_bf16(a, bg0f, cg0);
    cu0 = wmma_bf16(a, bu0f, cu0);
    cg1 = wmma_bf16(a, bg1f, cg1);
    cu1 = wmma_bf16(a, bu1f, cu1);
    __syncthreads();
  }

  // act = silu(gate) * up, guarded to valid rows
#pragma unroll
  for (int s = 0; s < 2; ++s) {
    const int ni = nj * 2 + s;
    v8f g = s ? cg1 : cg0;
    v8f u = s ? cu1 : cu0;
#pragma unroll
    for (int r = 0; r < 8; ++r) {
      int gm = M0 + mi * 16 + r + half * 8;
      if (gm < count) {
        float gv = g[r];
        float av = (gv / (1.f + __expf(-gv))) * u[r];
        act[(size_t)(base + gm) * FFN + n0 + ni * 16 + nlo] = f2bf(av);
      }
    }
  }
}

// ---------------- kernel 5: grouped GEMM2 (down proj) -> fp32 y --------------
__global__ __launch_bounds__(256) void k_gemm2(
    const u16* __restrict__ act, const float* __restrict__ wd,
    const int* __restrict__ cnt, const int* __restrict__ off,
    float* __restrict__ y) {
  const int e = blockIdx.z;
  const int count = cnt[e];
  const int M0 = blockIdx.y * 64;
  if (M0 >= count) return;
  const int base = off[e];
  const int n0 = blockIdx.x * 64;

  __shared__ u16 lA[64 * LSTR];
  __shared__ u16 lB[64 * LSTR];

  const int tid  = threadIdx.x;
  const int lane = tid & 31;
  const int wv   = tid >> 5;
  const int mi   = wv & 3;
  const int nj   = wv >> 2;

  v8f c0 = {}, c1 = {};

#if USE_TDM
  const u32 ldsA = lds_off(lA);
  const u32 rowsAvail = (u32)(NSLOT - (base + M0));
  const u16* atile = act + (size_t)(base + M0) * FFN;
#else
  const int ar = tid >> 2;
  const int ac = (tid & 3) * 8;
  int arow = base + M0 + ar; if (arow > NSLOT - 1) arow = NSLOT - 1;
  const u16* asrc = act + (size_t)arow * FFN + ac;
#endif

  const int bk = tid >> 3;
  const int bn = (tid & 7) * 8;
  const size_t wbase = (size_t)e * FFN * HIDDEN;

  const int nlo  = lane & 15;
  const int half = lane >> 4;

  for (int kk = 0; kk < FFN; kk += 32) {
#if USE_TDM
    if (wv == 0) tdm_load_tile(atile + kk, ldsA, FFN, rowsAvail);
#else
    *(uint4*)(lA + ar * LSTR + ac) = *(const uint4*)(asrc + kk);
#endif

    const float* bp = wd + wbase + (size_t)(kk + bk) * HIDDEN + (n0 + bn);
    if (kk + 32 < FFN) __builtin_prefetch(bp + (size_t)32 * HIDDEN, 0, 1);
    float4 b0 = *(const float4*)bp, b1 = *(const float4*)(bp + 4);
    u16* pb = lB + bk;
    pb[(bn + 0) * LSTR] = f2bf(b0.x); pb[(bn + 1) * LSTR] = f2bf(b0.y);
    pb[(bn + 2) * LSTR] = f2bf(b0.z); pb[(bn + 3) * LSTR] = f2bf(b0.w);
    pb[(bn + 4) * LSTR] = f2bf(b1.x); pb[(bn + 5) * LSTR] = f2bf(b1.y);
    pb[(bn + 6) * LSTR] = f2bf(b1.z); pb[(bn + 7) * LSTR] = f2bf(b1.w);

#if USE_TDM
    if (wv == 0) __builtin_amdgcn_s_wait_tensorcnt(0);
#endif
    __syncthreads();

    const u16* Ab  = lA + (mi * 16 + nlo) * LSTR + half * 16;
    const u16* Bp0 = lB + ((nj * 2) * 16 + nlo) * LSTR + half * 16;
    const u16* Bp1 = lB + ((nj * 2 + 1) * 16 + nlo) * LSTR + half * 16;
    v16bf a   = ld_frag(Ab, Ab + 8);
    v16bf b0f = ld_frag(Bp0, Bp0 + 8);
    v16bf b1f = ld_frag(Bp1, Bp1 + 8);
    c0 = wmma_bf16(a, b0f, c0);
    c1 = wmma_bf16(a, b1f, c1);
    __syncthreads();
  }

#pragma unroll
  for (int s = 0; s < 2; ++s) {
    const int ni = nj * 2 + s;
    v8f c = s ? c1 : c0;
#pragma unroll
    for (int r = 0; r < 8; ++r) {
      int gm = M0 + mi * 16 + r + half * 8;
      if (gm < count)
        y[(size_t)(base + gm) * HIDDEN + n0 + ni * 16 + nlo] = c[r];
    }
  }
}

// ---------------- kernel 6: unpermute with probs ----------------
__global__ __launch_bounds__(256) void k_combine(
    const float* __restrict__ y, const int* __restrict__ tok_e,
    const int* __restrict__ tok_pos, const float* __restrict__ tok_w,
    const int* __restrict__ off, float* __restrict__ out) {
  int t = blockIdx.x;
  int s0 = off[tok_e[t * 2 + 0]] + tok_pos[t * 2 + 0];
  int s1 = off[tok_e[t * 2 + 1]] + tok_pos[t * 2 + 1];
  float w0 = tok_w[t * 2 + 0];
  float w1 = tok_w[t * 2 + 1];
  int h = threadIdx.x * 4;
  float4 a = *(const float4*)(y + (size_t)s0 * HIDDEN + h);
  float4 b = *(const float4*)(y + (size_t)s1 * HIDDEN + h);
  float4 o;
  o.x = w0 * a.x + w1 * b.x;
  o.y = w0 * a.y + w1 * b.y;
  o.z = w0 * a.z + w1 * b.z;
  o.w = w0 * a.w + w1 * b.w;
  *(float4*)(out + (size_t)t * HIDDEN + h) = o;
}

// ---------------- host launcher ----------------
extern "C" void kernel_launch(void* const* d_in, const int* in_sizes, int n_in,
                              void* d_out, int out_size, void* d_ws, size_t ws_size,
                              hipStream_t stream) {
  (void)in_sizes; (void)n_in; (void)out_size; (void)ws_size;
  const float* x   = (const float*)d_in[0];   // [2,1024,1024]
  const float* gw  = (const float*)d_in[1];   // [8,1024]
  const float* wgu = (const float*)d_in[2];   // [8,1024,7168]
  const float* wd  = (const float*)d_in[3];   // [8,3584,1024]

  float* out    = (float*)d_out;                       // [T,H]
  float* logits = out + (size_t)T_TOK * HIDDEN;        // [T,8]

  // workspace layout
  int*   cnt     = (int*)d_ws;                 // 8
  int*   off     = cnt + 8;                    // 8
  int*   tok_e   = off + 8;                    // 4096
  int*   tok_pos = tok_e + NSLOT;              // 4096
  float* tok_w   = (float*)(tok_pos + NSLOT);  // 4096
  u16*   xp      = (u16*)(tok_w + NSLOT);      // 4096*1024 bf16 (8 MB)
  u16*   act     = xp + (size_t)NSLOT * HIDDEN;        // 4096*3584 bf16 (28.7 MB)
  float* y       = (float*)(act + (size_t)NSLOT * FFN); // 4096*1024 f32 (16 MB)

  k_init   <<<1, 32, 0, stream>>>(cnt);
  k_router <<<T_TOK / 8, 256, 0, stream>>>(x, gw, logits, cnt, tok_e, tok_pos, tok_w);
  k_offsets<<<1, 32, 0, stream>>>(cnt, off);
  k_gather <<<NSLOT, 256, 0, stream>>>(x, tok_e, tok_pos, off, xp);

  dim3 g1(FFN / 64, 32, NE);     // worst case M tiles = 2048/64 = 32, early-exit on cnt
  k_gemm1  <<<g1, 256, 0, stream>>>(xp, wgu, cnt, off, act);
  dim3 g2(HIDDEN / 64, 32, NE);
  k_gemm2  <<<g2, 256, 0, stream>>>(act, wd, cnt, off, y);

  k_combine<<<T_TOK, 256, 0, stream>>>(y, tok_e, tok_pos, tok_w, off, out);
}
